// SparseCoding_16372415332582
// MI455X (gfx1250) — matrix operand order repaired and checked by reading, not compiled, not run on hardware
//
#include <hip/hip_runtime.h>

typedef __attribute__((ext_vector_type(16))) __bf16   v16bf;
typedef __attribute__((ext_vector_type(8)))  __bf16   v8bf;
typedef __attribute__((ext_vector_type(2)))  __bf16   bf16x2;
typedef __attribute__((ext_vector_type(8)))  float    v8f;
typedef __attribute__((ext_vector_type(8)))  unsigned vu8;
typedef __attribute__((ext_vector_type(4)))  unsigned vu4;

#define B_TOT    32768
#define D_IN     128
#define D_OUT    512
#define N_STEPS  10
#define WAVES    4
#define ROWS_PER_WAVE 16
#define ROWS_PER_WG   (WAVES * ROWS_PER_WAVE)   // 64 -> 512 workgroups
#define NT       (D_OUT / 16)                   // 32 d_out tiles

#define GAMMA_C   0.1f
#define RHO_C     0.9f
#define LR_C      0.001f
#define MOM_C     0.9f
#define EPS_C     1e-8f
#define ABS_EPS_C 1e-6f

#define SWZ_SWAPX16 0x401F   // ds_swizzle group-of-32: and=0x1F, or=0, xor=0x10

#define W2_ELEMS  (D_OUT * D_IN)   // W  row-major  (bf16) 128 KB : A-frags for GEMM2^T
#define W1_ELEMS  (D_IN * D_OUT)   // W^T row-major (bf16) 128 KB : A-frags for GEMM1^T
#define SMEM_BYTES ((W2_ELEMS + W1_ELEMS) * sizeof(__bf16))   // 256 KB

// ---- A fragment (16x32 bf16) from a row-major LDS matrix -------------------
__device__ __forceinline__ v16bf load_afrag(const __bf16* p) {
    const v8bf lo = *(const v8bf*)(p);
    const v8bf hi = *(const v8bf*)(p + 8);
    return __builtin_shufflevector(lo, hi, 0,1,2,3,4,5,6,7,8,9,10,11,12,13,14,15);
}

__device__ __forceinline__ unsigned pack2(float a, float b) {
    union { bf16x2 h; unsigned u; } t;
    t.h = bf16x2{(__bf16)a, (__bf16)b};
    return t.u;
}

// packed-bf16 state helpers (bf16 -> f32 is exact: shift by 16)
__device__ __forceinline__ v8f unpack8(vu4 p) {
    v8f r;
    #pragma unroll
    for (int j = 0; j < 4; ++j) {
        union { unsigned u; float f; } a, b;
        a.u = p[j] << 16;
        b.u = p[j] & 0xffff0000u;
        r[2*j]     = a.f;
        r[2*j + 1] = b.f;
    }
    return r;
}
__device__ __forceinline__ vu4 pack8(v8f v) {
    vu4 p;
    #pragma unroll
    for (int j = 0; j < 4; ++j) p[j] = pack2(v[2*j], v[2*j + 1]);
    return p;
}

// ---- B fragment (32x16 bf16) built in registers ----------------------------
// State "CT" layout: tile q, elem r on lane L = value(dim 16q + r + 8h, batch ln).
// Lane L's B window is the full 16 rows of tile (2kc + h):
//   rows 0..7  always come from an h=0 lane's pack of that tile,
//   rows 8..15 always from an h=1 lane's pack.
//   => lo = hb ? recv : A4 ; hi = hb ? B4 : recv ; send = hb ? A4 : B4.
__device__ __forceinline__ v16bf build_bfrag(v8f tlo, v8f thi, bool hb) {
    unsigned A4[4], B4[4], recv[4], lo[4], hi[4];
    #pragma unroll
    for (int j = 0; j < 4; ++j) {
        A4[j] = pack2(tlo[2*j], tlo[2*j + 1]);   // tile 2kc   (own half rows)
        B4[j] = pack2(thi[2*j], thi[2*j + 1]);   // tile 2kc+1 (own half rows)
    }
    #pragma unroll
    for (int j = 0; j < 4; ++j) {                // batch the swizzles
        const unsigned send = hb ? A4[j] : B4[j];
        recv[j] = (unsigned)__builtin_amdgcn_ds_swizzle((int)send, SWZ_SWAPX16);
    }
    #pragma unroll
    for (int j = 0; j < 4; ++j) {
        lo[j] = hb ? recv[j] : A4[j];            // window rows 0..7
        hi[j] = hb ? B4[j]   : recv[j];          // window rows 8..15
    }
    union { v16bf bf; vu8 u; } f;
    f.u = vu8{lo[0], lo[1], lo[2], lo[3], hi[0], hi[1], hi[2], hi[3]};
    return f.bf;
}

__global__ __launch_bounds__(WAVES * 32, 1)
void sc_ista_wmma(const float* __restrict__ inp,
                  const float* __restrict__ Wg,
                  float* __restrict__ out) {
    extern __shared__ __align__(16) __bf16 smem[];
    __bf16* lW2 = smem;              // [D_OUT][D_IN]
    __bf16* lW1 = lW2 + W2_ELEMS;    // [D_IN][D_OUT]

    const int tid  = threadIdx.x;
    const int lane = tid & 31;
    const int wave = tid >> 5;
    const int ln   = lane & 15;      // batch column owned by this lane
    const int h    = lane >> 4;      // half
    const bool hb  = (h != 0);

    // ---- one-time cooperative W load (both orientations, bf16) ----
    for (int i = tid; i < W2_ELEMS; i += WAVES * 32) {
        const int r = i >> 7;             // d_out
        const int c = i & (D_IN - 1);     // d_in
        const __bf16 w = (__bf16)Wg[i];
        lW2[i] = w;
        lW1[c * D_OUT + r] = w;
    }
    __syncthreads();

    const int    row0   = blockIdx.x * ROWS_PER_WG + wave * ROWS_PER_WAVE;
    const float* inRowL = inp + (size_t)(row0 + ln) * D_IN;   // this lane's row

    const v8f zero8 = {0.f,0.f,0.f,0.f,0.f,0.f,0.f,0.f};
    const vu4 zero4 = {0u,0u,0u,0u};

    // x: f32 CT layout (256 VGPRs); a1/a2: packed bf16 pairs (128 VGPRs)
    v8f xr[NT];
    vu4 a1p[NT], a2p[NT];
    #pragma unroll
    for (int t = 0; t < NT; ++t) { xr[t] = zero8; a1p[t] = zero4; a2p[t] = zero4; }

    #pragma unroll 1
    for (int s = 0; s < N_STEPS; ++s) {
        // ===== GEMM1^T: recon^T[128 x 16] = W^T[128 x 512] @ x^T[512 x 16]
        v8f racc[8];
        #pragma unroll
        for (int m = 0; m < 8; ++m) racc[m] = zero8;

        #pragma unroll
        for (int kc = 0; kc < 16; ++kc) {            // K = d_out, 32-wide chunks
            const v16bf xb = build_bfrag(xr[2*kc], xr[2*kc + 1], hb);
            #pragma unroll
            for (int m = 0; m < 8; ++m) {            // M = d_in tiles
                const v16bf wa = load_afrag(lW1 + (16*m + ln) * D_OUT + 32*kc + 16*h);
                racc[m] = __builtin_amdgcn_wmma_f32_16x16x32_bf16(
                    false, wa, false, xb, (short)0, racc[m], false, false);
            }
            __builtin_amdgcn_sched_barrier(0);       // bound live ranges: no spills
        }

        // ===== e^T = 2*(recon^T - in^T) fused into the B-fragment build
        v16bf eb[4];
        #pragma unroll
        for (int kc = 0; kc < 4; ++kc) {
            v8f e0, e1;
            {
                const float4 i0 = *(const float4*)(inRowL + 16*(2*kc) + 8*h);
                const float4 i1 = *(const float4*)(inRowL + 16*(2*kc) + 8*h + 4);
                const v8f iv = {i0.x, i0.y, i0.z, i0.w, i1.x, i1.y, i1.z, i1.w};
                e0 = 2.0f * (racc[2*kc] - iv);
            }
            {
                const float4 i0 = *(const float4*)(inRowL + 16*(2*kc + 1) + 8*h);
                const float4 i1 = *(const float4*)(inRowL + 16*(2*kc + 1) + 8*h + 4);
                const v8f iv = {i0.x, i0.y, i0.z, i0.w, i1.x, i1.y, i1.z, i1.w};
                e1 = 2.0f * (racc[2*kc + 1] - iv);
            }
            eb[kc] = build_bfrag(e0, e1, hb);
            __builtin_amdgcn_sched_barrier(0);
        }

        // ===== GEMM2^T: g^T[512 x 16] = W[512 x 128] @ e^T[128 x 16]
        //       + fused smooth-L1 / RMSProp / momentum update per tile
        #pragma unroll
        for (int t = 0; t < NT; ++t) {
            v8f g = zero8;
            #pragma unroll
            for (int kc = 0; kc < 4; ++kc) {
                const v16bf wa = load_afrag(lW2 + (16*t + ln) * D_IN + 32*kc + 16*h);
                g = __builtin_amdgcn_wmma_f32_16x16x32_bf16(
                    false, wa, false, eb[kc], (short)0, g, false, false);
            }
            const v8f xv  = xr[t];
            const v8f a1v = unpack8(a1p[t]);
            const v8f a2v = unpack8(a2p[t]);
            v8f l1;
            #pragma unroll
            for (int r = 0; r < 8; ++r) l1[r] = rsqrtf(xv[r]*xv[r] + ABS_EPS_C);
            const v8f gt  = g + GAMMA_C * (xv * l1);
            const v8f na1 = RHO_C * a1v + (1.0f - RHO_C) * (gt * gt);
            v8f rs;
            #pragma unroll
            for (int r = 0; r < 8; ++r) rs[r] = rsqrtf(na1[r] + EPS_C);
            const v8f upd = (LR_C * gt) * rs;
            const v8f na2 = MOM_C * a2v - upd;
            xr[t]  = xv + MOM_C * na2 - upd;
            a1p[t] = pack8(na1);
            a2p[t] = pack8(na2);
            __builtin_amdgcn_sched_barrier(0);       // bound live ranges: no spills
        }
    }

    // ---- write final codes: per-lane contiguous 16B stores ----
    float* outRowL = out + (size_t)(row0 + ln) * D_OUT;
    #pragma unroll
    for (int t = 0; t < NT; ++t) {
        const v8f xv = xr[t];
        *(float4*)(outRowL + 16*t + 8*h)     = float4{xv[0], xv[1], xv[2], xv[3]};
        *(float4*)(outRowL + 16*t + 8*h + 4) = float4{xv[4], xv[5], xv[6], xv[7]};
    }
}

extern "C" void kernel_launch(void* const* d_in, const int* in_sizes, int n_in,
                              void* d_out, int out_size, void* d_ws, size_t ws_size,
                              hipStream_t stream) {
    const float* inputs = (const float*)d_in[0];   // [32768 x 128] f32
    const float* W      = (const float*)d_in[1];   // [512 x 128]   f32
    float*       out    = (float*)d_out;           // [32768 x 512] f32

    dim3 grid(B_TOT / ROWS_PER_WG);   // 512
    dim3 block(WAVES * 32);           // 128 threads = 4 waves
    sc_ista_wmma<<<grid, block, SMEM_BYTES, stream>>>(inputs, W, out);
}
